// EGATLayer_48163763257364
// MI455X (gfx1250) — matrix operand order repaired and checked by reading, not compiled, not run on hardware
//
#include <hip/hip_runtime.h>
#include <hip/hip_bf16.h>

// ---------------------------------------------------------------- types ----
typedef __bf16 bf16;
typedef bf16  v16bf __attribute__((ext_vector_type(16)));
typedef float v8f   __attribute__((ext_vector_type(8)));

union Frag16 { v16bf v; uint4 q[2]; unsigned short s[16]; };

__device__ __forceinline__ unsigned short f2bfu(float f) {
  union { float f; unsigned u; } v; v.f = f;
  unsigned u = v.u;
  unsigned r = u + 0x7FFFu + ((u >> 16) & 1u);   // round-to-nearest-even
  return (unsigned short)(r >> 16);
}
__device__ __forceinline__ unsigned pack2bf(float lo, float hi) {
  return (unsigned)f2bfu(lo) | ((unsigned)f2bfu(hi) << 16);
}

// Problem sizes: B=4, N=128, M=2048, D=64
// ws layout (float offsets)
static constexpr size_t OFF_ECS  = 0;                       // 4*16384*64 (-> EcT4 in place)
static constexpr size_t OFF_ELS  = OFF_ECS + 4194304;       // 4*16384*64 (-> ElT4 in place)
static constexpr size_t OFF_ECT  = OFF_ELS + 4194304;       // 4*2048*64
static constexpr size_t OFF_ELT  = OFF_ECT + 524288;        // 4*2048*64
static constexpr size_t OFF_HT   = OFF_ELT + 524288;        // 4*128*64
static constexpr size_t OFF_HTA  = OFF_HT  + 32768;         // 4 x 512 : [C:a1, C:a2, L:a1, L:a2]
static constexpr size_t OFF_EB   = OFF_HTA + 2048;          // 4 x 8192: [C:b1, C:b2, L:b1, L:b2]
static constexpr size_t OFF_HB3  = OFF_EB  + 32768;         // 2 x 512
static constexpr size_t OFF_ROWN = OFF_HB3 + 1024;          // 128 ints
static constexpr size_t OFF_ROWE = OFF_ROWN + 128;          // 2048 ints
static constexpr size_t OFF_PN   = OFF_ROWE + 2048;         // 128 ints
static constexpr size_t OFF_PE   = OFF_PN  + 128;           // 2048 ints
static constexpr size_t OFF_HNC  = OFF_PE  + 2048;          // 4*128*64
static constexpr size_t OFF_HNL  = OFF_HNC + 32768;
static constexpr size_t OFF_HMC  = OFF_HNL + 32768;
static constexpr size_t OFF_HML  = OFF_HMC + 32768;

// output offsets (floats): H_new | E_C_new | E_L_new | Hm
static constexpr size_t OUT_EC = 32768;
static constexpr size_t OUT_EL = 32768 + 524288;
static constexpr size_t OUT_HM = 32768 + 524288 + 524288;

// ------------------------------------------------------------ big GEMM ----
// Ecs = ME @ E_C, Els = ME @ E_L (fused; one pass over 512MB ME).
// Block: 256 thr (8 waves), 128x128 output tile, bf16 WMMA 16x16x32, f32 acc.
// Software-pipelined: global loads for tile i+1 issued before the WMMAs of
// tile i, so HBM latency hides under matrix math + LDS traffic.
__global__ __launch_bounds__(256)
void big_gemm(const float* __restrict__ ME, const float* __restrict__ EC,
              const float* __restrict__ EL, float* __restrict__ Ecs,
              float* __restrict__ Els)
{
  __shared__ unsigned short lA[128][40];   // [row][k] bf16, +8 pad
  __shared__ unsigned short lB[128][40];   // [col][k] bf16 (B transposed)
  const int tid  = threadIdx.x;
  const int lane = tid & 31;
  const int wave = tid >> 5;
  const int b    = blockIdx.y;
  const int rowbase = blockIdx.x * 128;

  v8f acc[8];
  #pragma unroll
  for (int n = 0; n < 8; n++)
    #pragma unroll
    for (int e = 0; e < 8; e++) acc[n][e] = 0.f;

  const size_t mebase = ((size_t)b * 16384 + rowbase) * 2048;
  const int arI = tid >> 3;              // A row within pass group (0..31)
  const int akq = (tid & 7) * 4;         // A k quad
  const int bkr = tid >> 5;              // B k row within pass group (0..7)
  const int bcq = (tid & 31) * 4;        // B col quad
  const float* bsrc = (bcq < 64)
      ? (EC + ((size_t)b * 2048) * 64 + bcq)
      : (EL + ((size_t)b * 2048) * 64 + (bcq - 64));

  float4 fA[4], fB[4];
  auto loadT = [&](int k0) {
    #pragma unroll
    for (int p = 0; p < 4; p++)
      fA[p] = *(const float4*)(ME + mebase + (size_t)(p * 32 + arI) * 2048 + k0 + akq);
    #pragma unroll
    for (int p = 0; p < 4; p++)
      fB[p] = *(const float4*)(bsrc + (size_t)(k0 + p * 8 + bkr) * 64);
  };
  auto storeT = [&]() {
    #pragma unroll
    for (int p = 0; p < 4; p++) {
      uint2 pk; pk.x = pack2bf(fA[p].x, fA[p].y); pk.y = pack2bf(fA[p].z, fA[p].w);
      *(uint2*)&lA[p * 32 + arI][akq] = pk;
    }
    #pragma unroll
    for (int p = 0; p < 4; p++) {
      int kr = p * 8 + bkr;
      lB[bcq + 0][kr] = f2bfu(fB[p].x); lB[bcq + 1][kr] = f2bfu(fB[p].y);
      lB[bcq + 2][kr] = f2bfu(fB[p].z); lB[bcq + 3][kr] = f2bfu(fB[p].w);
    }
  };

  loadT(0);                                // prologue
  for (int k0 = 0; k0 < 2048; k0 += 32) {
    storeT();                              // convert tile i -> LDS
    __syncthreads();
    if (k0 + 32 < 2048) loadT(k0 + 32);    // issue tile i+1 loads early

    Frag16 a;
    const int arow = wave * 16 + (lane & 15);
    const int kh   = (lane >> 4) * 8;
    a.q[0] = *(const uint4*)&lA[arow][kh];
    a.q[1] = *(const uint4*)&lA[arow][kh + 16];

    #pragma unroll
    for (int nt = 0; nt < 8; nt++) {
      Frag16 bfr;
      const int bcol = nt * 16 + (lane & 15);
      const int bk   = (lane >> 4) * 8;
      bfr.q[0] = *(const uint4*)&lB[bcol][bk];
      bfr.q[1] = *(const uint4*)&lB[bcol][bk + 16];
      acc[nt] = __builtin_amdgcn_wmma_f32_16x16x32_bf16(
          false, a.v, false, bfr.v, (short)0, acc[nt], false, false);
    }
    __syncthreads();
  }

  const int colb = lane & 15;
  const int rofs = (lane >> 4) * 8;
  #pragma unroll
  for (int nt = 0; nt < 8; nt++) {
    int col = nt * 16 + colb;
    float* dst = (col < 64)
        ? (Ecs + (((size_t)b * 16384 + rowbase + wave * 16) * 64) + col)
        : (Els + (((size_t)b * 16384 + rowbase + wave * 16) * 64) + (col - 64));
    #pragma unroll
    for (int r = 0; r < 8; r++)
      dst[(size_t)(r + rofs) * 64] = acc[nt][r];
  }
}

// ------------------------------------------------ generic rows x64 @ W64 ----
// dst = src @ W + bias (rows multiple of 128). Safe when dst == src.
__global__ __launch_bounds__(256)
void transform64(const float* src, float* dst,
                 const float* __restrict__ W, const float* __restrict__ bias)
{
  __shared__ unsigned short lA[128][72];
  __shared__ unsigned short lW[64][72];
  const int tid  = threadIdx.x;
  const int lane = tid & 31;
  const int wave = tid >> 5;
  const size_t rowbase = (size_t)blockIdx.x * 128;

  #pragma unroll
  for (int pass = 0; pass < 4; pass++) {              // W: 64x64 -> lW[n][k]
    int kr = pass * 16 + (tid >> 4);
    int cq = (tid & 15) * 4;
    float4 f = *(const float4*)(W + kr * 64 + cq);
    lW[cq + 0][kr] = f2bfu(f.x); lW[cq + 1][kr] = f2bfu(f.y);
    lW[cq + 2][kr] = f2bfu(f.z); lW[cq + 3][kr] = f2bfu(f.w);
  }
  #pragma unroll
  for (int pass = 0; pass < 8; pass++) {              // A: 128x64 (packed b64)
    int r  = pass * 16 + (tid >> 4);
    int cq = (tid & 15) * 4;
    float4 f = *(const float4*)(src + (rowbase + r) * 64 + cq);
    uint2 pk; pk.x = pack2bf(f.x, f.y); pk.y = pack2bf(f.z, f.w);
    *(uint2*)&lA[r][cq] = pk;
  }
  __syncthreads();

  v8f acc[4];
  #pragma unroll
  for (int n = 0; n < 4; n++)
    #pragma unroll
    for (int e = 0; e < 8; e++) acc[n][e] = 0.f;

  const int arow = wave * 16 + (lane & 15);
  #pragma unroll
  for (int kt = 0; kt < 2; kt++) {
    Frag16 a;
    int kh = kt * 32 + (lane >> 4) * 8;
    a.q[0] = *(const uint4*)&lA[arow][kh];
    a.q[1] = *(const uint4*)&lA[arow][kh + 16];
    #pragma unroll
    for (int nt = 0; nt < 4; nt++) {
      Frag16 bfr;
      int bcol = nt * 16 + (lane & 15);
      int bk   = kt * 32 + (lane >> 4) * 8;
      bfr.q[0] = *(const uint4*)&lW[bcol][bk];
      bfr.q[1] = *(const uint4*)&lW[bcol][bk + 16];
      acc[nt] = __builtin_amdgcn_wmma_f32_16x16x32_bf16(
          false, a.v, false, bfr.v, (short)0, acc[nt], false, false);
    }
  }

  const int colb = lane & 15;
  const int rofs = (lane >> 4) * 8;
  #pragma unroll
  for (int nt = 0; nt < 4; nt++) {
    int col = nt * 16 + colb;
    float bv = bias[col];
    #pragma unroll
    for (int r = 0; r < 8; r++)
      dst[(rowbase + wave * 16 + r + rofs) * 64 + col] = acc[nt][r] + bv;
  }
}

// ---------------------------------------------------------- small preps ----
__global__ __launch_bounds__(256)
void prep_flags(const int* __restrict__ pni, const int* __restrict__ pei,
                const int* __restrict__ AH, int* __restrict__ pN,
                int* __restrict__ pE, int* __restrict__ rowN)
{
  const int tid = threadIdx.x;
  for (int i = tid; i < 2048; i += 256) pE[i] = 0;
  if (tid < 128) pN[tid] = 0;
  __syncthreads();
  if (tid < 128) { int v = pni[tid]; if (v >= 0 && v < 128)  pN[v] = 1; }
  if (tid < 64)  { int v = pei[tid]; if (v >= 0 && v < 2048) pE[v] = 1; }
  __syncthreads();
  if (tid < 128) {
    int any = 0;
    for (int j = 0; j < 128; j++) if (AH[tid * 128 + j] > 0) { any = 1; break; }
    rowN[tid] = (pN[tid] && any) ? 1 : 0;
  }
}

__global__ __launch_bounds__(256)
void row_e(const int* __restrict__ AE, const int* __restrict__ pE,
           int* __restrict__ rowE)
{
  int p = blockIdx.x * 256 + threadIdx.x;
  int any = 0;
  const int* r = AE + (size_t)p * 2048;
  for (int q = 0; q < 2048; q++) if (r[q] > 0) { any = 1; break; }
  rowE[p] = (pE[p] && any) ? 1 : 0;
}

// per-side scalar projections: Eb1/Eb2 over (B*M), Hta1/Hta2/Hb3 over (B*N)
__global__ __launch_bounds__(256)
void proj(const float* __restrict__ Et, const float* __restrict__ Ht,
          const float* __restrict__ H, const float* __restrict__ aa,
          const float* __restrict__ bb, float* __restrict__ Eb1,
          float* __restrict__ Eb2, float* __restrict__ Hta1,
          float* __restrict__ Hta2, float* __restrict__ Hb3)
{
  const int gid = blockIdx.x * 256 + threadIdx.x;     // 0..8191
  const float* row = Et + (size_t)gid * 64;
  float d1 = 0.f, d2 = 0.f;
  for (int k = 0; k < 64; k++) { d1 += row[k] * bb[k]; d2 += row[k] * bb[64 + k]; }
  Eb1[gid] = d1; Eb2[gid] = d2;
  if (gid < 512) {
    const float* hr = Ht + (size_t)gid * 64;
    const float* h0 = H  + (size_t)gid * 64;
    float t1 = 0.f, t2 = 0.f, t3 = 0.f;
    for (int k = 0; k < 64; k++) {
      t1 += hr[k] * aa[k]; t2 += hr[k] * aa[64 + k]; t3 += h0[k] * bb[128 + k];
    }
    Hta1[gid] = t1; Hta2[gid] = t2; Hb3[gid] = t3;
  }
}

// -------------------------------------------------- node-side attention ----
__global__ __launch_bounds__(128)
void node_attn(const float* __restrict__ E4, const float* __restrict__ Ht,
               const float* __restrict__ a3, const float* __restrict__ Hta1,
               const float* __restrict__ Hta2, const int* __restrict__ AH,
               const int* __restrict__ rowN, float* __restrict__ Hn,
               float* __restrict__ Hm)
{
  __shared__ float sc[128];
  __shared__ float red[128];
  const int j = threadIdx.x;
  const int i = blockIdx.x, b = blockIdx.y;
  const size_t e4base = ((size_t)b * 16384 + (size_t)i * 128) * 64;

  const float* er = E4 + e4base + (size_t)j * 64;
  float d3 = 0.f;
  #pragma unroll
  for (int k = 0; k < 64; k++) d3 += er[k] * a3[k];
  float s = Hta1[b * 128 + i] + Hta2[b * 128 + j] + d3;
  s = (s >= 0.f) ? s : 0.2f * s;                       // leaky_relu(0.2)
  sc[j] = (AH[i * 128 + j] > 0) ? s : -1.0e9f;
  __syncthreads();

  red[j] = sc[j]; __syncthreads();
  for (int off = 64; off > 0; off >>= 1) {
    if (j < off) red[j] = fmaxf(red[j], red[j + off]);
    __syncthreads();
  }
  float mx = red[0]; __syncthreads();
  float e = __expf(sc[j] - mx);
  red[j] = e; __syncthreads();
  for (int off = 64; off > 0; off >>= 1) {
    if (j < off) red[j] += red[j + off];
    __syncthreads();
  }
  float tot = red[0]; __syncthreads();
  sc[j] = e / tot;
  __syncthreads();

  const int d = j & 63, g = j >> 6;
  float hn = 0.f, hm = 0.f;
  for (int jj = 0; jj < 64; jj++) {
    int j2 = g * 64 + jj;
    float a  = sc[j2];
    float ht = Ht[((size_t)b * 128 + j2) * 64 + d];
    float e4 = E4[e4base + (size_t)j2 * 64 + d];
    hn += a * ht; hm += a * ht * e4;
  }
  red[j] = hn; __syncthreads();
  float hn2 = (g == 0) ? (red[d] + red[64 + d]) : 0.f;
  __syncthreads();
  red[j] = hm; __syncthreads();
  if (g == 0) {
    float hm2 = red[d] + red[64 + d];
    bool  rn  = rowN[i] != 0;
    float base = Ht[((size_t)b * 128 + i) * 64 + d];
    Hn[((size_t)b * 128 + i) * 64 + d] = rn ? hn2 : base;
    Hm[((size_t)b * 128 + i) * 64 + d] = rn ? hm2 : 0.f;
  }
}

// -------------------------------------------------- edge-side attention ----
// Tiled: block = 128 p-rows (8 waves x 16), two-pass online softmax,
// aggregation attn(16x2048) @ Et(2048x64) via bf16 WMMA, Et streamed in
// 32-q chunks through LDS (Et read once per block instead of once per row).
__device__ __forceinline__ int edge_idx(int s1p, int s1q, int* validOut) {
  bool ca = (s1p == s1q);
  bool cb = (s1p == s1q + 1);
  bool cc = (s1p + 1 == s1q);
  *validOut = (ca || cb || cc) ? 1 : 0;
  int idx = (ca || cb) ? s1p : (cc ? (s1p + 1) : 0);
  return idx > 127 ? 127 : idx;
}

__global__ __launch_bounds__(256)
void edge_attn_tiled(const float* __restrict__ Et, const float* __restrict__ Eb1,
                     const float* __restrict__ Eb2, const float* __restrict__ Hb3,
                     const int* __restrict__ AE, const int* __restrict__ rowE,
                     float* __restrict__ out)
{
  __shared__ unsigned short lE[64][40];   // Et chunk, [d][k] bf16
  __shared__ int   sAE[128][32];          // AE chunk for the 128 rows
  __shared__ float sEb2[32];
  __shared__ float sHb3[128];
  __shared__ float sEb1[128];
  __shared__ float sM[128];               // per-row softmax max
  __shared__ float sInv[128];             // per-row 1/sum

  const int tid  = threadIdx.x;
  const int lane = tid & 31;
  const int wave = tid >> 5;
  const int p0   = blockIdx.x * 128;
  const int b    = blockIdx.y;
  const float* Eb2b = Eb2 + b * 2048;
  const float* Hb3b = Hb3 + b * 128;

  if (tid < 128) {
    sHb3[tid] = Hb3b[tid];
    sEb1[tid] = Eb1[b * 2048 + p0 + tid];
  }

  // -------- pass 1: per-row online max & sum over 2048 scores (wave/row) ----
  for (int rr = 0; rr < 16; rr++) {
    const int rloc = wave * 16 + rr;
    const int prow = p0 + rloc;
    const float eb1 = Eb1[b * 2048 + prow];
    const int s1p = (prow / 127) * 128;
    float m = -3.0e38f, ssum = 0.f;
    for (int it = 0; it < 64; it++) {
      int q = lane + it * 32;
      int s1q = (q / 127) * 128;
      int valid; int idx = edge_idx(s1p, s1q, &valid);
      float ht = valid ? Hb3b[idx] : 0.f;
      float s = eb1 + Eb2b[q] + ht;
      s = (s >= 0.f) ? s : 0.2f * s;
      s = (AE[(size_t)prow * 2048 + q] > 0) ? s : -1.0e9f;
      float mn = fmaxf(m, s);
      ssum = ssum * __expf(m - mn) + __expf(s - mn);
      m = mn;
    }
    // wave-level combine of (m, ssum)
    #pragma unroll
    for (int off = 16; off > 0; off >>= 1) {
      float m2 = __shfl_xor(m, off, 32);
      float s2 = __shfl_xor(ssum, off, 32);
      float mn = fmaxf(m, m2);
      ssum = ssum * __expf(m - mn) + s2 * __expf(m2 - mn);
      m = mn;
    }
    if (lane == 0) { sM[rloc] = m; sInv[rloc] = 1.f / ssum; }
  }
  __syncthreads();

  // -------- pass 2: stream q-chunks of 32, rebuild attn frags, WMMA ----------
  v8f acc[4];
  #pragma unroll
  for (int n = 0; n < 4; n++)
    #pragma unroll
    for (int e = 0; e < 8; e++) acc[n][e] = 0.f;

  const int arl = lane & 15;                 // row within wave's 16
  const int akh = (lane >> 4) * 8;           // A-frag k-half base

  for (int q0 = 0; q0 < 2048; q0 += 32) {
    // stage Et chunk (32 q x 64 d) transposed -> lE[d][k]
    #pragma unroll
    for (int pass = 0; pass < 2; pass++) {
      int lin = tid + pass * 256;            // 0..511
      int kr  = lin >> 4;
      int cq  = (lin & 15) * 4;
      float4 f = *(const float4*)(Et + ((size_t)b * 2048 + q0 + kr) * 64 + cq);
      lE[cq + 0][kr] = f2bfu(f.x); lE[cq + 1][kr] = f2bfu(f.y);
      lE[cq + 2][kr] = f2bfu(f.z); lE[cq + 3][kr] = f2bfu(f.w);
    }
    // stage AE chunk (128 rows x 32 q)
    #pragma unroll
    for (int pass = 0; pass < 4; pass++) {
      int lin = tid + pass * 256;            // 0..1023
      int r   = lin >> 3;
      int c4  = (lin & 7) * 4;
      *(int4*)&sAE[r][c4] = *(const int4*)(AE + (size_t)(p0 + r) * 2048 + q0 + c4);
    }
    if (tid < 32) sEb2[tid] = Eb2b[q0 + tid];
    __syncthreads();

    // build A-frag (attention weights, bf16) for this wave's 16 rows
    Frag16 a;
    {
      const int rloc = wave * 16 + arl;
      const int prow = p0 + rloc;
      const int s1p  = (prow / 127) * 128;
      const float eb1 = sEb1[rloc];
      const float m   = sM[rloc];
      const float inv = sInv[rloc];
      #pragma unroll
      for (int e = 0; e < 16; e++) {
        int kk = (e < 8) ? (akh + e) : (akh + 8 + e);   // {kh..kh+7, kh+16..kh+23}
        int q  = q0 + kk;
        int s1q = (q / 127) * 128;
        int valid; int idx = edge_idx(s1p, s1q, &valid);
        float ht = valid ? sHb3[idx] : 0.f;
        float s = eb1 + sEb2[kk] + ht;
        s = (s >= 0.f) ? s : 0.2f * s;
        s = (sAE[rloc][kk] > 0) ? s : -1.0e9f;
        a.s[e] = f2bfu(__expf(s - m) * inv);
      }
    }
    #pragma unroll
    for (int nt = 0; nt < 4; nt++) {
      Frag16 bfr;
      int bcol = nt * 16 + (lane & 15);
      int bk   = (lane >> 4) * 8;
      bfr.q[0] = *(const uint4*)&lE[bcol][bk];
      bfr.q[1] = *(const uint4*)&lE[bcol][bk + 16];
      acc[nt] = __builtin_amdgcn_wmma_f32_16x16x32_bf16(
          false, a.v, false, bfr.v, (short)0, acc[nt], false, false);
    }
    __syncthreads();
  }

  // -------- epilogue: rowE select, write out --------------------------------
  const int colb = lane & 15;
  const int rofs = (lane >> 4) * 8;
  #pragma unroll
  for (int nt = 0; nt < 4; nt++) {
    int col = nt * 16 + colb;
    #pragma unroll
    for (int r = 0; r < 8; r++) {
      int prow = p0 + wave * 16 + r + rofs;
      bool  re   = rowE[prow] != 0;
      float base = Et[((size_t)b * 2048 + prow) * 64 + col];
      out[((size_t)b * 2048 + prow) * 64 + col] = re ? acc[nt][r] : base;
    }
  }
}

__global__ __launch_bounds__(256)
void combine(const float* __restrict__ HnC, const float* __restrict__ HnL,
             const float* __restrict__ HmC, const float* __restrict__ HmL,
             float* __restrict__ out)
{
  int gid = blockIdx.x * 256 + threadIdx.x;            // 0..32767
  out[gid]          = 0.5f * (HnC[gid] + HnL[gid]);
  out[OUT_HM + gid] = 0.5f * (HmC[gid] + HmL[gid]);
}

// ---------------------------------------------------------------- host ----
extern "C" void kernel_launch(void* const* d_in, const int* in_sizes, int n_in,
                              void* d_out, int out_size, void* d_ws, size_t ws_size,
                              hipStream_t stream)
{
  const float* H    = (const float*)d_in[0];
  const float* E_C  = (const float*)d_in[1];
  const float* E_L  = (const float*)d_in[2];
  const int*   AH   = (const int*)d_in[3];
  const int*   AE   = (const int*)d_in[4];
  const float* ME   = (const float*)d_in[5];
  const int*   pni  = (const int*)d_in[7];
  const int*   pei  = (const int*)d_in[8];
  const float* W_H  = (const float*)d_in[9];
  const float* b_H  = (const float*)d_in[10];
  const float* W_EC = (const float*)d_in[11];
  const float* b_EC = (const float*)d_in[12];
  const float* W_EL = (const float*)d_in[13];
  const float* b_EL = (const float*)d_in[14];
  const float* a_C  = (const float*)d_in[15];
  const float* a_L  = (const float*)d_in[16];
  const float* b_C  = (const float*)d_in[17];
  const float* b_L  = (const float*)d_in[18];

  float* ws  = (float*)d_ws;
  float* out = (float*)d_out;

  float* wsEcs = ws + OFF_ECS;   // later EcT4
  float* wsEls = ws + OFF_ELS;   // later ElT4
  float* wsEcT = ws + OFF_ECT;
  float* wsElT = ws + OFF_ELT;
  float* wsHt  = ws + OFF_HT;
  int*   pN    = (int*)(ws + OFF_PN);
  int*   pE    = (int*)(ws + OFF_PE);
  int*   rowN  = (int*)(ws + OFF_ROWN);
  int*   rowE  = (int*)(ws + OFF_ROWE);

  // masks / path flags
  prep_flags<<<1, 256, 0, stream>>>(pni, pei, AH, pN, pE, rowN);
  row_e<<<8, 256, 0, stream>>>(AE, pE, rowE);

  // fused big GEMM: one pass over ME, bf16 WMMA, software pipelined
  big_gemm<<<dim3(128, 4), 256, 0, stream>>>(ME, E_C, E_L, wsEcs, wsEls);

  // linear transforms (WMMA): EcT4/ElT4 in place, EcT, ElT, Ht
  transform64<<<512, 256, 0, stream>>>(wsEcs, wsEcs, W_EC, b_EC);
  transform64<<<512, 256, 0, stream>>>(wsEls, wsEls, W_EL, b_EL);
  transform64<<<64, 256, 0, stream>>>(E_C, wsEcT, W_EC, b_EC);
  transform64<<<64, 256, 0, stream>>>(E_L, wsElT, W_EL, b_EL);
  transform64<<<4, 256, 0, stream>>>(H, wsHt, W_H, b_H);

  // scalar projections (per side)
  proj<<<32, 256, 0, stream>>>(wsEcT, wsHt, H, a_C, b_C,
                               ws + OFF_EB,          ws + OFF_EB + 8192,
                               ws + OFF_HTA,         ws + OFF_HTA + 512,
                               ws + OFF_HB3);
  proj<<<32, 256, 0, stream>>>(wsElT, wsHt, H, a_L, b_L,
                               ws + OFF_EB + 16384,  ws + OFF_EB + 24576,
                               ws + OFF_HTA + 1024,  ws + OFF_HTA + 1536,
                               ws + OFF_HB3 + 512);

  // node attention (side C, side L), then combine into H_new / Hm outputs
  node_attn<<<dim3(128, 4), 128, 0, stream>>>(wsEcs, wsHt, a_C + 128,
      ws + OFF_HTA, ws + OFF_HTA + 512, AH, rowN, ws + OFF_HNC, ws + OFF_HMC);
  node_attn<<<dim3(128, 4), 128, 0, stream>>>(wsEls, wsHt, a_L + 128,
      ws + OFF_HTA + 1024, ws + OFF_HTA + 1536, AH, rowN, ws + OFF_HNL, ws + OFF_HML);
  combine<<<128, 256, 0, stream>>>(ws + OFF_HNC, ws + OFF_HNL,
                                   ws + OFF_HMC, ws + OFF_HML, out);

  // edge attention, tiled + WMMA aggregation, direct to outputs
  edge_attn_tiled<<<dim3(16, 4), 256, 0, stream>>>(wsEcT,
      ws + OFF_EB,         ws + OFF_EB + 8192,  ws + OFF_HB3,       AE, rowE,
      out + OUT_EC);
  edge_attn_tiled<<<dim3(16, 4), 256, 0, stream>>>(wsElT,
      ws + OFF_EB + 16384, ws + OFF_EB + 24576, ws + OFF_HB3 + 512, AE, rowE,
      out + OUT_EL);
}
